// TransformerBlock_71184787964114
// MI455X (gfx1250) — matrix operand order repaired and checked by reading, not compiled, not run on hardware
//
#include <hip/hip_runtime.h>
#include <hip/hip_bf16.h>
#include <math.h>

// ---------------- problem constants ----------------
#define IMG     256
#define BATCH   4
#define CCH     256
#define HEADS   8
#define HDIM    32
#define SHIFT   4
#define NTOK    64
#define MTOT    (BATCH*IMG*IMG)     // 262144 tokens (== total window rows)
#define NWIN    (MTOT/NTOK)         // 4096 windows
#define MLPH    1024
#define ATT_SCALE 0.17677669529663687f   // 32^-0.5

typedef __attribute__((ext_vector_type(16))) __bf16 v16bf;
typedef __attribute__((ext_vector_type(8)))  float  v8f;

union Frag { v16bf v; unsigned u[8]; };

static __device__ __forceinline__ v8f vzero8() {
  v8f r = {0.f, 0.f, 0.f, 0.f, 0.f, 0.f, 0.f, 0.f};
  return r;
}

static __device__ __forceinline__ unsigned short f2bf(float f) {
  unsigned u = __builtin_bit_cast(unsigned, f);
  u = (u + 0x7FFFu + ((u >> 16) & 1u)) >> 16;   // round-to-nearest-even
  return (unsigned short)u;
}

// window-row index t -> source/dest pixel index (folds roll(+/-SHIFT) + window (de)partition)
static __device__ __forceinline__ int winrow_to_pixel(int t) {
  int win = t >> 6, tok = t & 63;
  int b  = win >> 10, rem = win & 1023;
  int wh = rem >> 5,  ww  = rem & 31;
  int i  = tok >> 3,  jj  = tok & 7;
  int h = (wh * 8 + i  + SHIFT) & (IMG - 1);
  int w = (ww * 8 + jj + SHIFT) & (IMG - 1);
  return (b * IMG + h) * IMG + w;
}

// async copy of 16 bytes/lane from global to LDS (GV mode, ASYNCcnt-tracked)
static __device__ __forceinline__ void async_b128(unsigned lds_off,
                                                  const void* gptr) {
  unsigned long long ga = (unsigned long long)gptr;
  asm volatile("global_load_async_to_lds_b128 %0, %1, off"
               :: "v"(lds_off), "v"(ga) : "memory");
}
static __device__ __forceinline__ void wait_async0() {
  asm volatile("s_wait_asynccnt 0" ::: "memory");
}

// ---------------- fp32 [K,N] -> bf16 transposed [N,K] weight conversion ----------------
__global__ __launch_bounds__(256) void cvt_t_kernel(const float* __restrict__ in,
                                                    unsigned short* __restrict__ out,
                                                    int K, int N) {
  int i = blockIdx.x * 256 + threadIdx.x;
  if (i < K * N) {
    int k = i / N, n = i - k * N;
    out[(size_t)n * K + k] = f2bf(in[i]);
  }
}

// ---------------- LayerNorm (optionally fused with shift+window gather) ----------------
// one wave (32 lanes) per token row of 256 channels; 8 tokens per 256-thread block
__global__ __launch_bounds__(256) void ln_kernel(const float* __restrict__ x,
                                                 const float* __restrict__ g,
                                                 const float* __restrict__ b,
                                                 unsigned short* __restrict__ out,
                                                 int gather) {
  int wave = threadIdx.x >> 5, lane = threadIdx.x & 31;
  int t = (blockIdx.x << 3) + wave;
  int src = gather ? winrow_to_pixel(t) : t;
  const float* row = x + (size_t)src * CCH;
  float vals[8];
  float s = 0.f;
  #pragma unroll
  for (int i = 0; i < 8; ++i) { vals[i] = row[lane + (i << 5)]; s += vals[i]; }
  #pragma unroll
  for (int m = 1; m < 32; m <<= 1) s += __shfl_xor(s, m);
  float mean = s * (1.f / CCH);
  float var = 0.f;
  #pragma unroll
  for (int i = 0; i < 8; ++i) { float d = vals[i] - mean; var += d * d; }
  #pragma unroll
  for (int m = 1; m < 32; m <<= 1) var += __shfl_xor(var, m);
  float rstd = rsqrtf(var * (1.f / CCH) + 1e-5f);
  size_t ob = (size_t)t * CCH;
  #pragma unroll
  for (int i = 0; i < 8; ++i) {
    int c = lane + (i << 5);
    out[ob + c] = f2bf((vals[i] - mean) * rstd * g[c] + b[c]);
  }
}

// ---------------- motion-bias MLP: mb[win][head][tok] ----------------
__global__ __launch_bounds__(256) void mb_kernel(const float* __restrict__ flow,
                                                 const float* __restrict__ w1,
                                                 const float* __restrict__ b1,
                                                 const float* __restrict__ w2,
                                                 const float* __restrict__ b2,
                                                 float* __restrict__ mb) {
  int t = blockIdx.x * 256 + threadIdx.x;
  int p = winrow_to_pixel(t);
  float f0 = flow[2 * (size_t)p], f1 = flow[2 * (size_t)p + 1];
  float o[8];
  #pragma unroll
  for (int h = 0; h < 8; ++h) o[h] = b2[h];
  for (int j = 0; j < 32; ++j) {
    float hj = fmaxf(f0 * w1[j] + f1 * w1[32 + j] + b1[j], 0.f);
    #pragma unroll
    for (int h = 0; h < 8; ++h) o[h] += hj * w2[j * 8 + h];
  }
  int win = t >> 6, tok = t & 63;
  #pragma unroll
  for (int h = 0; h < 8; ++h)
    mb[(((size_t)win * 8 + h) << 6) + tok] = o[h];
}

// ---------------- tiled bf16 WMMA GEMM, async-LDS double-buffered ----------------
// C[M,N] = A[M,K] * Bt[N,K]^T (+fused epilogues)
// block tile 128x128, 8 waves, wave tile 32x64 (2x4 accum tiles), BK = 32
// A and Bt tiles are both k-contiguous 64B rows -> staged with
// global_load_async_to_lds_b128 (ASYNCcnt) into double-buffered LDS.
// mode 0: outH = bf16(acc + bias)                       (QKV)
// mode 1: outH = bf16(gelu(acc + bias))                 (MLP fc1)
// mode 2: outF[pix(row)] = acc + bias + resid[pix(row)] (proj + window-reverse + roll + shortcut)
// mode 3: outF[row] += acc + bias                       (MLP fc2 + residual, in place in d_out)
__global__ __launch_bounds__(256) void gemm_bf16_wmma(
    const unsigned short* __restrict__ A,    // [M,K] bf16
    const unsigned short* __restrict__ Bt,   // [N,K] bf16 (pre-transposed weights)
    const float* __restrict__ bias,
    int M, int N, int K, int mode,
    const float* __restrict__ resid,
    float* __restrict__ outF,
    unsigned short* __restrict__ outH) {
  __shared__ __align__(16) unsigned short As[2][128][32];   // [buf][m][k]
  __shared__ __align__(16) unsigned short Bs[2][128][32];   // [buf][n][k]

  const int tid  = threadIdx.x;
  const int lane = tid & 31, wave = tid >> 5;
  const int wm = wave & 3, wn = wave >> 2;
  const int lr = lane & 15, lh = lane >> 4;
  const int rowBase = blockIdx.x * 128;
  const int colBase = blockIdx.y * 128;

  // stage one 128x32 A tile + 128x32 B tile into LDS buffer `buf` (async)
  auto stage = [&](int kt, int buf) {
    #pragma unroll
    for (int i = 0; i < 2; ++i) {
      int q = tid + (i << 8);          // 512 chunks of 16B per tile
      int r = q >> 2, c8 = (q & 3) << 3;
      async_b128((unsigned)(unsigned long long)&As[buf][r][c8],
                 A  + (size_t)(rowBase + r) * K + kt + c8);
      async_b128((unsigned)(unsigned long long)&Bs[buf][r][c8],
                 Bt + (size_t)(colBase + r) * K + kt + c8);
    }
  };

  v8f acc[2][4];
  #pragma unroll
  for (int mt = 0; mt < 2; ++mt)
    #pragma unroll
    for (int nt = 0; nt < 4; ++nt) acc[mt][nt] = vzero8();

  int cur = 0;
  stage(0, 0);
  wait_async0();
  __syncthreads();

  for (int kt = 0; kt < K; kt += 32) {
    // prefetch next k-tile into the other buffer while we compute this one
    if (kt + 32 < K) stage(kt + 32, cur ^ 1);

    // A fragments: lane holds row m; k = (v>=4?16:0) + (lane>=16?8:0) + 2(v&3)+h
    Frag af[2];
    #pragma unroll
    for (int mt = 0; mt < 2; ++mt) {
      int ml = wm * 32 + mt * 16 + lr;
      #pragma unroll
      for (int v = 0; v < 8; ++v) {
        int k = ((v & 4) << 2) + (lh << 3) + ((v & 3) << 1);
        af[mt].u[v] = *(const unsigned*)&As[cur][ml][k];
      }
    }
    // B fragments: lane holds col n; k = (v>=4?16:0) + (lane>=16?8:0) + 2(v&3)+h
    Frag bfr[4];
    #pragma unroll
    for (int nt = 0; nt < 4; ++nt) {
      int nl = wn * 64 + nt * 16 + lr;
      #pragma unroll
      for (int v = 0; v < 8; ++v) {
        int k = ((v & 4) << 2) + (lh << 3) + ((v & 3) << 1);
        bfr[nt].u[v] = *(const unsigned*)&Bs[cur][nl][k];
      }
    }
    #pragma unroll
    for (int mt = 0; mt < 2; ++mt)
      #pragma unroll
      for (int nt = 0; nt < 4; ++nt)
        acc[mt][nt] = __builtin_amdgcn_wmma_f32_16x16x32_bf16(
            false, af[mt].v, false, bfr[nt].v, (short)0, acc[mt][nt], false, false);

    wait_async0();        // next tile's async copies have landed in LDS
    __syncthreads();      // everyone done reading `cur`, writes visible
    cur ^= 1;
  }

  // fused epilogue; D layout: col = base + (lane&15); row = base + r + (lane>=16?8:0)
  #pragma unroll
  for (int mt = 0; mt < 2; ++mt)
    #pragma unroll
    for (int nt = 0; nt < 4; ++nt) {
      int col = colBase + wn * 64 + nt * 16 + lr;
      float bc = bias[col];
      #pragma unroll
      for (int r = 0; r < 8; ++r) {
        int row = rowBase + wm * 32 + mt * 16 + r + (lh << 3);
        float val = acc[mt][nt][r] + bc;
        if (mode == 0) {
          outH[(size_t)row * N + col] = f2bf(val);
        } else if (mode == 1) {
          val = 0.5f * val * (1.0f + erff(val * 0.70710678118f));
          outH[(size_t)row * N + col] = f2bf(val);
        } else if (mode == 2) {
          int p = winrow_to_pixel(row);
          size_t idx = (size_t)p * N + col;
          outF[idx] = val + resid[idx];
        } else {
          size_t idx = (size_t)row * N + col;
          outF[idx] = outF[idx] + val;
        }
      }
    }
}

// ---------------- window attention: S = QK^T*s + mb, softmax, O = PV ----------------
// one block (4 waves) per (window, head); each wave owns 16 query rows
__global__ __launch_bounds__(128) void attn_wmma_kernel(
    const unsigned short* __restrict__ qkv,   // [MTOT, 768] bf16; q|k|v at col 0|256|512
    const float* __restrict__ mb,             // [NWIN*HEADS*64]
    unsigned short* __restrict__ outw) {      // [MTOT, 256] bf16, window order
  const int bid = blockIdx.x;
  const int win = bid >> 3, hd = bid & 7;
  const int lane = threadIdx.x & 31, wave = threadIdx.x >> 5;
  const int lr = lane & 15, lh = lane >> 4;

  __shared__ __align__(16) unsigned short Pl[64][64];   // probabilities, bf16
  __shared__ __align__(16) unsigned short Vt[32][64];   // V transposed: [d][key]

  const unsigned short* base = qkv + (size_t)win * 64 * 768;

  // stage V transposed so P*V B-fragments read contiguous key pairs
  for (int idx = threadIdx.x; idx < 2048; idx += 128) {
    int key = idx >> 5, d = idx & 31;
    Vt[d][key] = base[(size_t)key * 768 + 512 + hd * 32 + d];
  }

  // Q A-fragment straight from global (row = query, packed d pairs)
  const int qrow = wave * 16 + lr;
  Frag qf;
  #pragma unroll
  for (int v = 0; v < 8; ++v) {
    int d = ((v & 4) << 2) + (lh << 3) + ((v & 3) << 1);
    qf.u[v] = *(const unsigned*)(base + (size_t)qrow * 768 + hd * 32 + d);
  }

  // S = Q K^T : 4 key tiles of 16
  v8f s[4];
  #pragma unroll
  for (int j = 0; j < 4; ++j) {
    Frag kf;
    int key = j * 16 + lr;
    #pragma unroll
    for (int v = 0; v < 8; ++v) {
      int d = ((v & 4) << 2) + (lh << 3) + ((v & 3) << 1);
      kf.u[v] = *(const unsigned*)(base + (size_t)key * 768 + 256 + hd * 32 + d);
    }
    s[j] = __builtin_amdgcn_wmma_f32_16x16x32_bf16(
        false, qf.v, false, kf.v, (short)0, vzero8(), false, false);
  }

  // scale + motion bias + row softmax over 64 keys (4 tiles x 16 lanes per half)
  const float* mrow = mb + (((size_t)win * 8 + hd) << 6);
  #pragma unroll
  for (int r = 0; r < 8; ++r) {
    int m = wave * 16 + r + (lh << 3);
    float qb = mrow[m];
    float best = -1e30f;
    #pragma unroll
    for (int j = 0; j < 4; ++j) {
      float v = s[j][r] * ATT_SCALE + qb;
      s[j][r] = v;
      best = fmaxf(best, v);
    }
    #pragma unroll
    for (int msk = 1; msk < 16; msk <<= 1) best = fmaxf(best, __shfl_xor(best, msk));
    float acc = 0.f;
    #pragma unroll
    for (int j = 0; j < 4; ++j) { float p = __expf(s[j][r] - best); s[j][r] = p; acc += p; }
    #pragma unroll
    for (int msk = 1; msk < 16; msk <<= 1) acc += __shfl_xor(acc, msk);
    float inv = 1.f / acc;
    #pragma unroll
    for (int j = 0; j < 4; ++j)
      Pl[m][j * 16 + lr] = f2bf(s[j][r] * inv);
  }
  __syncthreads();

  // O = P V : K=64 split in two 32-chunks, 2 output d-tiles
  v8f o[2] = { vzero8(), vzero8() };
  #pragma unroll
  for (int kc = 0; kc < 2; ++kc) {
    Frag pf;
    #pragma unroll
    for (int v = 0; v < 8; ++v) {
      int k = kc * 32 + ((v & 4) << 2) + (lh << 3) + ((v & 3) << 1);
      pf.u[v] = *(const unsigned*)&Pl[qrow][k];
    }
    #pragma unroll
    for (int nt = 0; nt < 2; ++nt) {
      Frag vf;
      int n = nt * 16 + lr;
      #pragma unroll
      for (int v = 0; v < 8; ++v) {
        int kk = kc * 32 + ((v & 4) << 2) + (lh << 3) + ((v & 3) << 1);
        vf.u[v] = *(const unsigned*)&Vt[n][kk];
      }
      o[nt] = __builtin_amdgcn_wmma_f32_16x16x32_bf16(
          false, pf.v, false, vf.v, (short)0, o[nt], false, false);
    }
  }

  #pragma unroll
  for (int nt = 0; nt < 2; ++nt)
    #pragma unroll
    for (int r = 0; r < 8; ++r) {
      int row = win * 64 + wave * 16 + r + (lh << 3);
      int col = hd * 32 + nt * 16 + lr;
      outw[(size_t)row * 256 + col] = f2bf(o[nt][r]);
    }
}

// ---------------- host orchestration ----------------
extern "C" void kernel_launch(void* const* d_in, const int* in_sizes, int n_in,
                              void* d_out, int out_size, void* d_ws, size_t ws_size,
                              hipStream_t stream) {
  (void)in_sizes; (void)n_in; (void)out_size; (void)ws_size;
  const float* x      = (const float*)d_in[0];
  const float* flow   = (const float*)d_in[1];
  const float* qkv_w  = (const float*)d_in[2];
  const float* qkv_b  = (const float*)d_in[3];
  const float* proj_w = (const float*)d_in[4];
  const float* proj_b = (const float*)d_in[5];
  const float* off_w1 = (const float*)d_in[6];
  const float* off_b1 = (const float*)d_in[7];
  const float* off_w2 = (const float*)d_in[8];
  const float* off_b2 = (const float*)d_in[9];
  const float* n1_g   = (const float*)d_in[10];
  const float* n1_b   = (const float*)d_in[11];
  const float* n2_g   = (const float*)d_in[12];
  const float* n2_b   = (const float*)d_in[13];
  const float* mlp_w1 = (const float*)d_in[14];
  const float* mlp_b1 = (const float*)d_in[15];
  const float* mlp_w2 = (const float*)d_in[16];
  const float* mlp_b2 = (const float*)d_in[17];
  float* out = (float*)d_out;

  // workspace carve-up (all offsets 256B aligned)
  char* w = (char*)d_ws;
  size_t off = 0;
  unsigned short* bigH = (unsigned short*)(w + off); off += (size_t)MTOT * MLPH * 2; // qkv, later mlp hidden
  unsigned short* xnH  = (unsigned short*)(w + off); off += (size_t)MTOT * CCH * 2;  // ln1 out, later ln2 out
  unsigned short* aoH  = (unsigned short*)(w + off); off += (size_t)MTOT * CCH * 2;  // attention out (window order)
  float* mbF           = (float*)(w + off);          off += (size_t)NWIN * HEADS * NTOK * 4;
  unsigned short* qkvWH  = (unsigned short*)(w + off); off += 196608ULL * 2;  // [768,256]
  unsigned short* projWH = (unsigned short*)(w + off); off += 65536ULL * 2;   // [256,256]
  unsigned short* w1H    = (unsigned short*)(w + off); off += 262144ULL * 2;  // [1024,256]
  unsigned short* w2H    = (unsigned short*)(w + off); off += 262144ULL * 2;  // [256,1024]

  // weights -> bf16, transposed to [N,K] so GEMM B tiles are k-contiguous
  cvt_t_kernel<<<(196608 + 255) / 256, 256, 0, stream>>>(qkv_w, qkvWH, 256, 768);
  cvt_t_kernel<<<(65536  + 255) / 256, 256, 0, stream>>>(proj_w, projWH, 256, 256);
  cvt_t_kernel<<<(262144 + 255) / 256, 256, 0, stream>>>(mlp_w1, w1H, 256, 1024);
  cvt_t_kernel<<<(262144 + 255) / 256, 256, 0, stream>>>(mlp_w2, w2H, 1024, 256);

  // LN1 fused with shift + window partition (window-ordered bf16 rows)
  ln_kernel<<<MTOT / 8, 256, 0, stream>>>(x, n1_g, n1_b, xnH, 1);
  // motion bias MLP
  mb_kernel<<<MTOT / 256, 256, 0, stream>>>(flow, off_w1, off_b1, off_w2, off_b2, mbF);

  // QKV GEMM: [262144,256] x [256,768]
  gemm_bf16_wmma<<<dim3(MTOT / 128, 768 / 128), 256, 0, stream>>>(
      xnH, qkvWH, qkv_b, MTOT, 768, 256, 0, nullptr, nullptr, bigH);

  // window attention
  attn_wmma_kernel<<<NWIN * HEADS, 128, 0, stream>>>(bigH, mbF, aoH);

  // proj GEMM + window-reverse + roll + shortcut residual -> y in d_out (f32)
  gemm_bf16_wmma<<<dim3(MTOT / 128, 256 / 128), 256, 0, stream>>>(
      aoH, projWH, proj_b, MTOT, 256, 256, 2, x, out, nullptr);

  // LN2 on y (plain order)
  ln_kernel<<<MTOT / 8, 256, 0, stream>>>(out, n2_g, n2_b, xnH, 0);

  // MLP fc1 + GELU: [262144,256] x [256,1024]
  gemm_bf16_wmma<<<dim3(MTOT / 128, 1024 / 128), 256, 0, stream>>>(
      xnH, w1H, mlp_b1, MTOT, 1024, 256, 1, nullptr, nullptr, bigH);

  // MLP fc2 + residual in place: d_out = y + h
  gemm_bf16_wmma<<<dim3(MTOT / 128, 256 / 128), 256, 0, stream>>>(
      bigH, w2H, mlp_b2, MTOT, 256, 1024, 3, nullptr, out, nullptr);
}
